// TGSR_26147760898594
// MI455X (gfx1250) — matrix operand order repaired and checked by reading, not compiled, not run on hardware
//
#include <hip/hip_runtime.h>
#include <stdint.h>

// ---------------------------------------------------------------------------
// y_t = M (L y_{t-1} + x_t),  M = inv(L+I)  ==>  y_t = P y_{t-1} + Z_t
//   P = M L (once),  Z = M xs (batched GEMM written into d_out),
//   scan: out[t] = P*out[t-1] + out[t]  (127 sequential small GEMMs)
// inv via Newton-Schulz (all-GEMM, f32 WMMA); main path bf16 WMMA.
// Slice staging uses the Tensor Data Mover (tensor_load_to_lds + TENSORcnt).
// ---------------------------------------------------------------------------

typedef __attribute__((ext_vector_type(2)))  float        v2f;
typedef __attribute__((ext_vector_type(8)))  float        v8f;
typedef __attribute__((ext_vector_type(8)))  __bf16       v8bf;
typedef __attribute__((ext_vector_type(16))) __bf16       v16bf;
typedef __attribute__((ext_vector_type(4)))  unsigned int v4u;
typedef __attribute__((ext_vector_type(8)))  int          v8i;
typedef __attribute__((ext_vector_type(4)))  int          v4i;

#define NMAT 512           // node count (square matrices are 512x512)
#define FDIM 64            // feature dim
#define SLICE (NMAT*FDIM)  // 32768 elems per (b,t) slice
#define LDB 520            // padded LDS row stride (bf16) for transposed staging
#define LDST_BYTES ((size_t)FDIM * LDB * sizeof(__bf16))   // 66560
#define CHUNK 16384        // f32 bounce chunk: 256 nodes x 64 feats (64KB)
#define LDS_TOTAL (LDST_BYTES + CHUNK * sizeof(float))     // 132096 B

// ---------------- row/col sums of A -----------------------------------------
__global__ __launch_bounds__(256) void k_sums(const float* __restrict__ A,
                                              float* __restrict__ rowsum,
                                              float* __restrict__ colsum) {
  __shared__ float red[256];
  int id = blockIdx.x, tid = threadIdx.x;
  float s = 0.f;
  if (id < NMAT) {
    const float* r = A + (size_t)id * NMAT;
    for (int j = tid; j < NMAT; j += 256) s += r[j];
  } else {
    int j = id - NMAT;
    for (int i = tid; i < NMAT; i += 256) s += A[(size_t)i * NMAT + j];
  }
  red[tid] = s; __syncthreads();
  for (int off = 128; off > 0; off >>= 1) {
    if (tid < off) red[tid] += red[tid + off];
    __syncthreads();
  }
  if (tid == 0) { if (id < NMAT) rowsum[id] = red[0]; else colsum[id - NMAT] = red[0]; }
}

// ---------------- L = D - A ;  G = L + I ------------------------------------
__global__ __launch_bounds__(256) void k_build(const float* __restrict__ A,
                                               const float* __restrict__ rowsum,
                                               float* __restrict__ G,
                                               float* __restrict__ L) {
  int i = blockIdx.x, tid = threadIdx.x;
  float Di = rowsum[i];
  for (int j = tid; j < NMAT; j += 256) {
    float a = A[(size_t)i * NMAT + j];
    float l = (i == j ? Di : 0.f) - a;
    L[(size_t)i * NMAT + j] = l;
    G[(size_t)i * NMAT + j] = l + (i == j ? 1.f : 0.f);
  }
}

// ---------------- X0 = G^T / (||G||_1 ||G||_inf) ----------------------------
__global__ __launch_bounds__(256) void k_x0(const float* __restrict__ A,
                                            const float* __restrict__ rowsum,
                                            const float* __restrict__ colsum,
                                            const float* __restrict__ G,
                                            float* __restrict__ X0) {
  __shared__ float mr[256], mc[256];
  int i = blockIdx.x, tid = threadIdx.x;
  float r = 0.f, c = 0.f;
  for (int j = tid; j < NMAT; j += 256) {
    float d  = A[(size_t)j * NMAT + j];
    float ra = 1.f + 2.f * (rowsum[j] - d);               // abs row sum of G
    float ca = colsum[j] + rowsum[j] + 1.f - 2.f * d;     // abs col sum of G
    r = fmaxf(r, ra); c = fmaxf(c, ca);
  }
  mr[tid] = r; mc[tid] = c; __syncthreads();
  for (int off = 128; off > 0; off >>= 1) {
    if (tid < off) { mr[tid] = fmaxf(mr[tid], mr[tid + off]);
                     mc[tid] = fmaxf(mc[tid], mc[tid + off]); }
    __syncthreads();
  }
  float alpha = 1.f / (mr[0] * mc[0]);
  for (int j = tid; j < NMAT; j += 256)
    X0[(size_t)i * NMAT + j] = alpha * G[(size_t)j * NMAT + i];
}

// ---------------- f32 WMMA GEMM, 512x512x512 --------------------------------
// MODE 0: C = Aop @ Bop     MODE 1: C = 2*Dold - Aop @ Bop  (Newton-Schulz)
template <int MODE>
__global__ __launch_bounds__(256) void k_gemm_f32(float* __restrict__ C,
                                                  const float* __restrict__ Aop,
                                                  const float* __restrict__ Bop,
                                                  const float* __restrict__ Dold) {
  int tid = threadIdx.x;
  int lane = tid & 31, wave = tid >> 5;
  int tile = blockIdx.x * 8 + wave;        // 1024 tiles of 16x16
  int tm = tile >> 5, tn = tile & 31;
  int lm = lane & 15, hi = lane >> 4;
  int n = tn * 16 + lm;
  const float* arow = Aop + (size_t)(tm * 16 + lm) * NMAT;
  v8f acc = {};
  for (int k = 0; k < NMAT; k += 4) {
    v2f a = *(const v2f*)(arow + k + 2 * hi);            // A: K = {2hi, 2hi+1}
    v2f b;                                               // B: VGPR r -> K = r+2hi
    b[0] = Bop[(size_t)(k + 2 * hi + 0) * NMAT + n];
    b[1] = Bop[(size_t)(k + 2 * hi + 1) * NMAT + n];
    acc = __builtin_amdgcn_wmma_f32_16x16x4_f32(false, a, false, b,
                                                (short)0, acc, false, false);
  }
  int mbase = tm * 16 + hi * 8;
  if (MODE == 1) {
    float d[8];
#pragma unroll
    for (int r = 0; r < 8; ++r) d[r] = Dold[(size_t)(mbase + r) * NMAT + n];
#pragma unroll
    for (int r = 0; r < 8; ++r)
      C[(size_t)(mbase + r) * NMAT + n] = 2.f * d[r] - acc[r];
  } else {
#pragma unroll
    for (int r = 0; r < 8; ++r)
      C[(size_t)(mbase + r) * NMAT + n] = acc[r];
  }
}

// ---------------- f32 -> bf16 conversion ------------------------------------
__global__ __launch_bounds__(256) void k_cvt(const float* __restrict__ src,
                                             __bf16* __restrict__ dst, int n) {
  int i = blockIdx.x * 256 + threadIdx.x;
  if (i < n) dst[i] = (__bf16)src[i];
}

// ---------------- bf16 fragment loaders (ISA 7.12.2 layouts) ----------------
// A 16x32 bf16: lane m, elems e<8 -> K=k0+8*hi+e ; e>=8 -> K=k0+16+8*hi+(e-8)
__device__ inline v16bf load_afrag(const __bf16* __restrict__ Ab, int m, int k0,
                                   int hi) {
  const __bf16* p = Ab + (size_t)m * NMAT + k0 + 8 * hi;
  v8bf r0 = *(const v8bf*)(p);
  v8bf r1 = *(const v8bf*)(p + 16);
  return __builtin_shufflevector(r0, r1, 0,1,2,3,4,5,6,7,8,9,10,11,12,13,14,15);
}
// B 32x16 bf16 from LDS staged TRANSPOSED as [n][k] (stride LDB):
// lane col n, elem e<8 -> K = k0 + 8*hi + e ; e>=8 -> K = k0 + 16 + 8*hi + (e-8)
__device__ inline v16bf load_bfrag(const __bf16* __restrict__ lds, int n, int k0,
                                   int hi) {
  const __bf16* p = lds + (size_t)n * LDB + k0 + 8 * hi;
  v8bf r0 = *(const v8bf*)(p);
  v8bf r1 = *(const v8bf*)(p + 16);
  return __builtin_shufflevector(r0, r1, 0,1,2,3,4,5,6,7,8,9,10,11,12,13,14,15);
}

// ---------------- TDM: DMA `elems` f32 (contiguous) from global into LDS ----
// D# per ISA 8.3/8.4: group0 = {count=1, lds_addr, global_addr, type=2},
// group1 = {data_size=4B, tensor_dim0=elems, tile_dim0=elems, stride0=elems}.
// 6-arg builtin (clang-23 arity): (v4u, v8i, v4i, v4i, v8i, i32 cpol).
__device__ inline void tdm_load_f32(const float* __restrict__ src,
                                    const float* lds_dst, int elems) {
  unsigned long long ga = (unsigned long long)(uintptr_t)src;
  unsigned la = (unsigned)(uintptr_t)lds_dst;          // flat LDS -> low 32 bits
  v4u g0; v8i g1; v4i gz4; v8i gz8;
  g0[0] = 1u;                                          // count=1 (valid), user D#
  g0[1] = la;                                          // lds_addr (bytes)
  g0[2] = (unsigned)(ga & 0xFFFFFFFFu);                // global_addr[31:0]
  g0[3] = (unsigned)((ga >> 32) & 0x01FFFFFFu)         // global_addr[56:32]
          | 0x80000000u;                               // type=2 ("image")
  g1[0] = (int)(2u << 16);                             // data_size = 2 -> 4 bytes
  g1[1] = (int)((unsigned)(elems & 0xFFFF) << 16);     // tensor_dim0[15:0]
  g1[2] = (int)((unsigned)elems >> 16);                // tensor_dim0[31:16]
  g1[3] = (int)((unsigned)(elems & 0xFFFF) << 16);     // tile_dim0
  g1[4] = 0;                                           // tile_dim1/2 unused
  g1[5] = elems;                                       // tensor_dim0_stride[31:0]
  g1[6] = 0;
  g1[7] = 0;
  gz4[0] = 0; gz4[1] = 0; gz4[2] = 0; gz4[3] = 0;      // groups 2/3 unused (1-D)
#pragma unroll
  for (int q = 0; q < 8; ++q) gz8[q] = 0;
  __builtin_amdgcn_tensor_load_to_lds(g0, g1, gz4, gz4, gz8, 0);
}

// Stage one [512 nodes x 64 feats] f32 slice -> transposed bf16 LDS [64][LDB].
// Two rounds: TDM DMAs 64KB f32 into bounce buffer, threads convert+transpose.
__device__ inline void stage_slice(const float* __restrict__ src,
                                   __bf16* __restrict__ ldsT,
                                   float* __restrict__ ldsF, int tid) {
  int wave = tid >> 5;
  for (int round = 0; round < 2; ++round) {
    // each of 8 waves DMAs 2048 f32 (8KB) of this round's 16384-elem chunk
    tdm_load_f32(src + round * CHUNK + wave * (CHUNK / 8),
                 ldsF + wave * (CHUNK / 8), CHUNK / 8);
    __builtin_amdgcn_s_wait_tensorcnt(0);
    __syncthreads();                                   // all waves' DMA visible
    for (int s = tid; s < CHUNK; s += 256) {
      int kl = s >> 6, n = s & 63;
      ldsT[n * LDB + (round * 256 + kl)] = (__bf16)ldsF[s];
    }
    __syncthreads();                                   // done reading bounce buf
  }
}

// ---------------- Z = M @ xs  (one block per (b,t) slice) -------------------
__global__ __launch_bounds__(256) void k_zmul(float* __restrict__ out,
                                              const float* __restrict__ xs,
                                              const __bf16* __restrict__ Mbf) {
  extern __shared__ char smem[];
  __bf16* ldsT = (__bf16*)smem;                        // [FDIM][LDB] transposed
  float*  ldsF = (float*)(smem + LDST_BYTES);          // f32 bounce chunk
  size_t bt = blockIdx.x;
  const float* src = xs + bt * SLICE;
  float* dst = out + bt * SLICE;
  int tid = threadIdx.x;
  stage_slice(src, ldsT, ldsF, tid);
  int lane = tid & 31, wave = tid >> 5;
  int lm = lane & 15, hi = lane >> 4;
  for (int i = 0; i < 16; ++i) {                       // 128 tiles / 8 waves
    int tile = wave * 16 + i;
    int tm = tile >> 2, tn = tile & 3;
    v8f acc = {};
    for (int k0 = 0; k0 < NMAT; k0 += 32) {
      v16bf a = load_afrag(Mbf, tm * 16 + lm, k0, hi);
      v16bf b = load_bfrag(ldsT, tn * 16 + lm, k0, hi);
      acc = __builtin_amdgcn_wmma_f32_16x16x32_bf16(false, a, false, b,
                                                    (short)0, acc, false, false);
    }
#pragma unroll
    for (int r = 0; r < 8; ++r) {
      int mm = tm * 16 + hi * 8 + r;
      dst[mm * FDIM + tn * 16 + lm] = acc[r];
    }
  }
}

// ---------------- scan step: out[t] = P @ out[t-1] + out[t] -----------------
__global__ __launch_bounds__(256) void k_step(float* __restrict__ out,
                                              const __bf16* __restrict__ Pbf,
                                              int t) {
  extern __shared__ char smem[];
  __bf16* ldsT = (__bf16*)smem;
  float*  ldsF = (float*)(smem + LDST_BYTES);
  int b = blockIdx.x >> 3, rg = blockIdx.x & 7;        // 8 batches x 8 row groups
  const float* yp = out + ((size_t)b * 128 + (t - 1)) * SLICE;
  float*       yc = out + ((size_t)b * 128 + t) * SLICE;
  int tid = threadIdx.x;
  stage_slice(yp, ldsT, ldsF, tid);
  int lane = tid & 31, wave = tid >> 5;
  int lm = lane & 15, hi = lane >> 4;
  for (int i = 0; i < 2; ++i) {                        // 16 tiles / 8 waves
    int tile = wave * 2 + i;
    int tml = tile >> 2, tn = tile & 3;
    int m0 = rg * 64 + tml * 16;
    v8f acc = {};
    for (int k0 = 0; k0 < NMAT; k0 += 32) {
      v16bf a = load_afrag(Pbf, m0 + lm, k0, hi);
      v16bf b = load_bfrag(ldsT, tn * 16 + lm, k0, hi);
      acc = __builtin_amdgcn_wmma_f32_16x16x32_bf16(false, a, false, b,
                                                    (short)0, acc, false, false);
    }
#pragma unroll
    for (int r = 0; r < 8; ++r) {
      int mm = m0 + hi * 8 + r;
      size_t idx = (size_t)mm * FDIM + tn * 16 + lm;
      yc[idx] = acc[r] + yc[idx];                      // += Z_t (already there)
    }
  }
}

// ---------------------------------------------------------------------------
extern "C" void kernel_launch(void* const* d_in, const int* in_sizes, int n_in,
                              void* d_out, int out_size, void* d_ws, size_t ws_size,
                              hipStream_t stream) {
  const float* xs = (const float*)d_in[0];   // [8,128,512,64]
  const float* A  = (const float*)d_in[1];   // [1,512,512]
  float* out = (float*)d_out;                // [8,128,512,64]

  // workspace carve (~7.4 MB)
  float* wf      = (float*)d_ws;
  float* rowsum  = wf;
  float* colsum  = wf + 512;
  float* G       = wf + 1024;
  float* L       = G  + NMAT * NMAT;
  float* X0      = L  + NMAT * NMAT;
  float* X1      = X0 + NMAT * NMAT;
  float* R       = X1 + NMAT * NMAT;
  float* Pf      = R  + NMAT * NMAT;
  __bf16* Mbf    = (__bf16*)(Pf + NMAT * NMAT);
  __bf16* Pbf    = Mbf + NMAT * NMAT;

  k_sums <<<2 * NMAT, 256, 0, stream>>>(A, rowsum, colsum);
  k_build<<<NMAT,     256, 0, stream>>>(A, rowsum, G, L);
  k_x0   <<<NMAT,     256, 0, stream>>>(A, rowsum, colsum, G, X0);

  // Newton-Schulz: X <- 2X - X(GX), 32 iters (quadratic; G diag-dominant)
  float* Xc = X0; float* Xn = X1;
  for (int it = 0; it < 32; ++it) {
    k_gemm_f32<0><<<128, 256, 0, stream>>>(R,  G,  Xc, nullptr);  // R = G X
    k_gemm_f32<1><<<128, 256, 0, stream>>>(Xn, Xc, R,  Xc);       // X = 2X - X R
    float* tmp = Xc; Xc = Xn; Xn = tmp;
  }
  k_gemm_f32<0><<<128, 256, 0, stream>>>(Pf, Xc, L, nullptr);     // P = M L

  k_cvt<<<(NMAT * NMAT + 255) / 256, 256, 0, stream>>>(Xc, Mbf, NMAT * NMAT);
  k_cvt<<<(NMAT * NMAT + 255) / 256, 256, 0, stream>>>(Pf, Pbf, NMAT * NMAT);

  k_zmul<<<1024, 256, LDS_TOTAL, stream>>>(out, xs, Mbf);         // Z -> d_out
  for (int t = 1; t < 128; ++t)                                   // sequential scan
    k_step<<<64, 256, LDS_TOTAL, stream>>>(out, Pbf, t);
}
